// MultiHeadSelfAttention_36567351558305
// MI455X (gfx1250) — matrix operand order repaired and checked by reading, not compiled, not run on hardware
//
#include <hip/hip_runtime.h>
#include <hip/hip_bf16.h>

typedef __attribute__((ext_vector_type(16))) _Float16 v16h;
typedef __attribute__((ext_vector_type(8)))  _Float16 v8h;
typedef __attribute__((ext_vector_type(8)))  float    v8f;

#define BB 2
#define LL 2048
#define DD 512
#define HH 8
#define CW 4096  // H*D comb width

#define BM 128
#define BN 128
#define BK 32
#define BKP (BK + 8)
#define BNP (BN + 8)
#define NTHREADS 256

struct alignas(16) H16 { _Float16 v[16]; };

// ---- 16-element row load (+ fp32 -> fp16 convert) ------------------------
__device__ __forceinline__ void ld16cvt(_Float16* dst, const float* src) {
#pragma unroll
  for (int i = 0; i < 16; i += 4) {
    float4 t = *(const float4*)(src + i);
    dst[i + 0] = (_Float16)t.x;
    dst[i + 1] = (_Float16)t.y;
    dst[i + 2] = (_Float16)t.z;
    dst[i + 3] = (_Float16)t.w;
  }
}
__device__ __forceinline__ void ld16cvt(_Float16* dst, const _Float16* src) {
  *(v8h*)(dst)     = *(const v8h*)(src);
  *(v8h*)(dst + 8) = *(const v8h*)(src + 8);
}

// ---- shared GEMM core: C[128x128] block, f32 accum, f16 WMMA --------------
// NN only: A [M,K] row-major (f32 or f16), B [K,N] row-major f16.
// Double-buffered LDS (ping-pong), one barrier per K-step, global prefetch
// into registers issued before the WMMA block of the current step.
template <typename AT>
__device__ __forceinline__ void gemm_core(const AT* __restrict__ A, int lda,
                                          const _Float16* __restrict__ Bmat, int ldb,
                                          int K, int mblk, int nblk,
                                          v8f (&acc)[4][2]) {
  __shared__ __align__(16) _Float16 As[2][BM][BKP];
  __shared__ __align__(16) _Float16 Bs[2][BK][BNP];

  const int tid  = threadIdx.x;
  const int lane = tid & 31;
  const int wave = tid >> 5;
  const int wm   = (wave >> 2) * 64;  // wave m-origin in block tile
  const int wn   = (wave & 3) * 32;   // wave n-origin in block tile
  const int half = lane >> 4;
  const int lm   = lane & 15;

  const int ar = tid >> 1;        // 0..127 (row for A staging)
  const int ac = (tid & 1) * 16;  // 0 or 16 (k segment)
  const int kr = tid >> 3;        // 0..31  (k row for B staging)
  const int nc = (tid & 7) * 16;  // n segment for B staging

  const v8f vzero = {};
#pragma unroll
  for (int i = 0; i < 4; i++)
#pragma unroll
    for (int j = 0; j < 2; j++) acc[i][j] = vzero;

  H16 ra, rb;

  // ---- prologue: stage K-tile 0 into buffer 0 ----
  ld16cvt(ra.v, A + (size_t)(mblk + ar) * lda + ac);
  ld16cvt(rb.v, Bmat + (size_t)kr * ldb + (nblk + nc));
  *(v8h*)&As[0][ar][ac]     = *(const v8h*)&ra.v[0];
  *(v8h*)&As[0][ar][ac + 8] = *(const v8h*)&ra.v[8];
  *(v8h*)&Bs[0][kr][nc]     = *(const v8h*)&rb.v[0];
  *(v8h*)&Bs[0][kr][nc + 8] = *(const v8h*)&rb.v[8];
  __syncthreads();

  const int niter = K / BK;
  for (int it = 0; it < niter; ++it) {
    const int cur = it & 1;

    // ---- prefetch next K-tile from global into registers ----
    if (it + 1 < niter) {
      const int k0 = (it + 1) * BK;
      ld16cvt(ra.v, A + (size_t)(mblk + ar) * lda + (k0 + ac));
      ld16cvt(rb.v, Bmat + (size_t)(k0 + kr) * ldb + (nblk + nc));
    }

    // ---- fragment loads per documented gfx1250 WMMA layouts ----
    v16h af[4], bf[2];
#pragma unroll
    for (int i = 0; i < 4; i++) {
      const v8h a0 = *(const v8h*)&As[cur][wm + i * 16 + lm][half * 8];
      const v8h a1 = *(const v8h*)&As[cur][wm + i * 16 + lm][16 + half * 8];
#pragma unroll
      for (int e = 0; e < 8; e++) { af[i][e] = a0[e]; af[i][e + 8] = a1[e]; }
    }
#pragma unroll
    for (int j = 0; j < 2; j++) {
#pragma unroll
      for (int e = 0; e < 8; e++) {
        bf[j][e]     = Bs[cur][half * 8 + e][wn + j * 16 + lm];
        bf[j][e + 8] = Bs[cur][16 + half * 8 + e][wn + j * 16 + lm];
      }
    }

#pragma unroll
    for (int i = 0; i < 4; i++)
#pragma unroll
      for (int j = 0; j < 2; j++)
        acc[i][j] = __builtin_amdgcn_wmma_f32_16x16x32_f16(
            false, af[i], false, bf[j], (short)0, acc[i][j], false, false);

    // ---- commit prefetched tile into the alternate buffer ----
    if (it + 1 < niter) {
      const int nxt = cur ^ 1;
      *(v8h*)&As[nxt][ar][ac]     = *(const v8h*)&ra.v[0];
      *(v8h*)&As[nxt][ar][ac + 8] = *(const v8h*)&ra.v[8];
      *(v8h*)&Bs[nxt][kr][nc]     = *(const v8h*)&rb.v[0];
      *(v8h*)&Bs[nxt][kr][nc + 8] = *(const v8h*)&rb.v[8];
      __syncthreads();  // one barrier per K-step
    }
  }
}

// ---- kernel 0: transpose + convert weights: f32 [R,C] -> f16 [C,R] --------
__global__ __launch_bounds__(NTHREADS) void cvtT_kernel(
    const float* __restrict__ src, _Float16* __restrict__ dst, int R, int C) {
  __shared__ _Float16 t[32][33];
  src += (size_t)blockIdx.z * R * C;
  dst += (size_t)blockIdx.z * R * C;
  const int tx = threadIdx.x & 31;
  const int ty = (threadIdx.x >> 5) * 4;  // 8 waves -> 4 rows each
  const int r0 = blockIdx.y * 32, c0 = blockIdx.x * 32;
#pragma unroll
  for (int i = 0; i < 4; i++)
    t[ty + i][tx] = (_Float16)src[(size_t)(r0 + ty + i) * C + (c0 + tx)];
  __syncthreads();
#pragma unroll
  for (int i = 0; i < 4; i++)
    dst[(size_t)(c0 + ty + i) * R + (r0 + tx)] = t[tx][ty + i];
}

// ---- kernel 1: per-head projections (48 batched NN GEMMs) -----------------
// qh, vh stored row-major [L,D]; kh stored TRANSPOSED khT [D,L] with
// vectorized 16B stores (8 contiguous m per lane).
__global__ __launch_bounds__(NTHREADS) void proj_kernel(
    const float* __restrict__ q, const float* __restrict__ k,
    const float* __restrict__ v, const _Float16* __restrict__ WqT,
    const _Float16* __restrict__ WkT, const _Float16* __restrict__ WvT,
    _Float16* __restrict__ qh, _Float16* __restrict__ khT,
    _Float16* __restrict__ vh) {
  const int z     = blockIdx.z;
  const int which = z >> 4;   // 0:q 1:k 2:v
  const int bh    = z & 15;
  const int b     = bh >> 3;
  const int h     = bh & 7;
  const float* A = (which == 0 ? q : which == 1 ? k : v) + (size_t)b * LL * DD;
  const _Float16* W =
      (which == 0 ? WqT : which == 1 ? WkT : WvT) + (size_t)h * DD * DD;

  v8f acc[4][2];
  gemm_core<float>(A, DD, W, DD, DD, blockIdx.x * BM, blockIdx.y * BN, acc);

  const int tid = threadIdx.x, lane = tid & 31, wave = tid >> 5;
  const int mb = blockIdx.x * BM + (wave >> 2) * 64 + (lane >> 4) * 8;
  const int nb = blockIdx.y * BN + (wave & 3) * 32 + (lane & 15);
  if (which == 1) {
    _Float16* out = khT + (size_t)bh * LL * DD;  // [D][L]
#pragma unroll
    for (int i = 0; i < 4; i++)
#pragma unroll
      for (int j = 0; j < 2; j++) {
        const int e = nb + j * 16;
        v8h h8;
#pragma unroll
        for (int r = 0; r < 8; r++) h8[r] = (_Float16)acc[i][j][r];
        *(v8h*)&out[(size_t)e * LL + (mb + i * 16)] = h8;  // 8 contiguous m
      }
  } else {
    _Float16* out = (which == 0 ? qh : vh) + (size_t)bh * LL * DD;  // [L][D]
#pragma unroll
    for (int i = 0; i < 4; i++)
#pragma unroll
      for (int j = 0; j < 2; j++)
#pragma unroll
        for (int r = 0; r < 8; r++)
          out[(size_t)(mb + i * 16 + r) * DD + (nb + j * 16)] =
              (_Float16)acc[i][j][r];
  }
}

// ---- kernel 2: scores = qh * khT / sqrt(D) (NN), fp32 into d_out ----------
__global__ __launch_bounds__(NTHREADS) void scores_kernel(
    const _Float16* __restrict__ qh, const _Float16* __restrict__ khT,
    float* __restrict__ attn) {
  const int bh = blockIdx.z;
  const _Float16* A  = qh + (size_t)bh * LL * DD;   // [L,D]
  const _Float16* Bp = khT + (size_t)bh * LL * DD;  // [D,L]
  float* out = attn + (size_t)bh * LL * LL;

  v8f acc[4][2];
  gemm_core<_Float16>(A, DD, Bp, LL, DD, blockIdx.x * BM, blockIdx.y * BN, acc);

  const float scale = 0.044194173824159216f;  // 1/sqrt(512)
  const int tid = threadIdx.x, lane = tid & 31, wave = tid >> 5;
  const int mb = blockIdx.x * BM + (wave >> 2) * 64 + (lane >> 4) * 8;
  const int nb = blockIdx.y * BN + (wave & 3) * 32 + (lane & 15);
#pragma unroll
  for (int i = 0; i < 4; i++)
#pragma unroll
    for (int j = 0; j < 2; j++)
#pragma unroll
      for (int r = 0; r < 8; r++)
        out[(size_t)(mb + i * 16 + r) * LL + (nb + j * 16)] = acc[i][j][r] * scale;
}

// ---- kernel 3: row softmax over L=2048, in place ---------------------------
__global__ __launch_bounds__(NTHREADS) void softmax_kernel(float* __restrict__ attn) {
  __shared__ float red[NTHREADS];
  const int t = threadIdx.x;
  float* row = attn + (size_t)blockIdx.x * LL;

  float x[8];
#pragma unroll
  for (int i = 0; i < 8; i++) x[i] = row[t * 8 + i];

  float m = x[0];
#pragma unroll
  for (int i = 1; i < 8; i++) m = fmaxf(m, x[i]);
  red[t] = m;
  __syncthreads();
  for (int s = NTHREADS / 2; s > 0; s >>= 1) {
    if (t < s) red[t] = fmaxf(red[t], red[t + s]);
    __syncthreads();
  }
  const float rmax = red[0];
  __syncthreads();

  float sum = 0.0f;
#pragma unroll
  for (int i = 0; i < 8; i++) {
    x[i] = __expf(x[i] - rmax);
    sum += x[i];
  }
  red[t] = sum;
  __syncthreads();
  for (int s = NTHREADS / 2; s > 0; s >>= 1) {
    if (t < s) red[t] += red[t + s];
    __syncthreads();
  }
  const float inv = 1.0f / red[0];
#pragma unroll
  for (int i = 0; i < 8; i++) row[t * 8 + i] = x[i] * inv;
}

// ---- kernel 4: ctx = attn * vh (NN), store f16 in permuted comb layout ----
__global__ __launch_bounds__(NTHREADS) void ctx_kernel(
    const float* __restrict__ attn, const _Float16* __restrict__ vh,
    _Float16* __restrict__ comb) {
  const int bh = blockIdx.z;
  const int b  = bh >> 3;
  const int h  = bh & 7;
  const float* A = attn + (size_t)bh * LL * LL;
  const _Float16* Bp = vh + (size_t)bh * LL * DD;
  _Float16* out = comb + (size_t)b * LL * CW;

  v8f acc[4][2];
  gemm_core<float>(A, LL, Bp, DD, LL, blockIdx.x * BM, blockIdx.y * BN, acc);

  const int tid = threadIdx.x, lane = tid & 31, wave = tid >> 5;
  const int mb = blockIdx.x * BM + (wave >> 2) * 64 + (lane >> 4) * 8;
  const int nb = blockIdx.y * BN + (wave & 3) * 32 + (lane & 15);
#pragma unroll
  for (int i = 0; i < 4; i++)
#pragma unroll
    for (int j = 0; j < 2; j++) {
      const int e    = nb + j * 16;                        // 0..511
      const int pcol = (e >> 6) * 512 + h * 64 + (e & 63); // comb permutation
#pragma unroll
      for (int r = 0; r < 8; r++)
        out[(size_t)(mb + i * 16 + r) * CW + pcol] = (_Float16)acc[i][j][r];
    }
}

// ---- kernel 5: out = comb * WoT (NN), fp32 --------------------------------
__global__ __launch_bounds__(NTHREADS) void out_kernel(
    const _Float16* __restrict__ comb, const _Float16* __restrict__ WoT,
    float* __restrict__ out) {
  const int b = blockIdx.z;
  const _Float16* A = comb + (size_t)b * LL * CW;  // [L, 4096]
  float* O = out + (size_t)b * LL * DD;

  v8f acc[4][2];
  gemm_core<_Float16>(A, CW, WoT, DD, CW, blockIdx.x * BM, blockIdx.y * BN, acc);

  const int tid = threadIdx.x, lane = tid & 31, wave = tid >> 5;
  const int mb = blockIdx.x * BM + (wave >> 2) * 64 + (lane >> 4) * 8;
  const int nb = blockIdx.y * BN + (wave & 3) * 32 + (lane & 15);
#pragma unroll
  for (int i = 0; i < 4; i++)
#pragma unroll
    for (int j = 0; j < 2; j++)
#pragma unroll
      for (int r = 0; r < 8; r++)
        O[(size_t)(mb + i * 16 + r) * DD + (nb + j * 16)] = acc[i][j][r];
}

extern "C" void kernel_launch(void* const* d_in, const int* in_sizes, int n_in,
                              void* d_out, int out_size, void* d_ws, size_t ws_size,
                              hipStream_t stream) {
  const float* q  = (const float*)d_in[0];
  const float* k  = (const float*)d_in[1];
  const float* v  = (const float*)d_in[2];
  const float* Wq = (const float*)d_in[3];
  const float* Wk = (const float*)d_in[4];
  const float* Wv = (const float*)d_in[5];
  const float* Wo = (const float*)d_in[6];

  float* out  = (float*)d_out;
  float* attn = out + (size_t)BB * LL * DD;  // d_out = [out | attn] flat

  const size_t headsz = (size_t)BB * HH * LL * DD;  // 16.7M f16 each
  const size_t wsz    = (size_t)HH * DD * DD;       // 2.1M f16 per W tensor
  _Float16* qh   = (_Float16*)d_ws;
  _Float16* khT  = qh + headsz;   // [bh][D][L] (transposed)
  _Float16* vh   = khT + headsz;
  _Float16* comb = vh + headsz;   // [B, L, 4096]
  _Float16* WqT  = comb + headsz; // [h][D(in d)][D(out e)]
  _Float16* WkT  = WqT + wsz;
  _Float16* WvT  = WkT + wsz;
  _Float16* WoT  = WvT + wsz;     // [4096, 512]

  dim3 blk(NTHREADS);
  // weight transpose+convert (once, amortized over all GEMM blocks)
  cvtT_kernel<<<dim3(DD / 32, DD / 32, HH), blk, 0, stream>>>(Wq, WqT, DD, DD);
  cvtT_kernel<<<dim3(DD / 32, DD / 32, HH), blk, 0, stream>>>(Wk, WkT, DD, DD);
  cvtT_kernel<<<dim3(DD / 32, DD / 32, HH), blk, 0, stream>>>(Wv, WvT, DD, DD);
  cvtT_kernel<<<dim3(CW / 32, DD / 32, 1), blk, 0, stream>>>(Wo, WoT, DD, CW);

  proj_kernel<<<dim3(LL / BM, DD / BN, 3 * BB * HH), blk, 0, stream>>>(
      q, k, v, WqT, WkT, WvT, qh, khT, vh);
  scores_kernel<<<dim3(LL / BM, LL / BN, BB * HH), blk, 0, stream>>>(qh, khT, attn);
  softmax_kernel<<<dim3(BB * HH * LL), blk, 0, stream>>>(attn);
  ctx_kernel<<<dim3(LL / BM, DD / BN, BB * HH), blk, 0, stream>>>(attn, vh, comb);
  out_kernel<<<dim3(LL / BM, DD / BN, BB), blk, 0, stream>>>(comb, WoT, out);
}
